// CochAV_75187697484004
// MI455X (gfx1250) — matrix operand order repaired and checked by reading, not compiled, not run on hardware
//
#include <hip/hip_runtime.h>
#include <hip/hip_bf16.h>
#include <math.h>

// ---------------------------------------------------------------------------
// CDNA5 (gfx1250) WMMA types & helpers.  wave32; D = A(16x32 bf16) * B(32x16
// bf16) + C(16x16 f32).  Layouts per CDNA5 ISA 7.12.2.
// ---------------------------------------------------------------------------
typedef __attribute__((ext_vector_type(16))) __bf16 v16bf;
typedef __attribute__((ext_vector_type(8)))  float  v8f;

#define WMMA_BF16(A, B, C) \
  __builtin_amdgcn_wmma_f32_16x16x32_bf16(false, (A), false, (B), (short)0, (C), false, false)

__device__ __forceinline__ unsigned short f2bf(float f) {
  const __bf16 h = (__bf16)f;                 // native v_cvt_pk_bf16_f32 path
  return __builtin_bit_cast(unsigned short, h);
}
__device__ __forceinline__ float bf2f(unsigned short h) {
  union { unsigned u; float f; } v; v.u = ((unsigned)h) << 16;
  return v.f;
}
__device__ __forceinline__ __bf16 bfbits(unsigned short h) {
  return __builtin_bit_cast(__bf16, h);
}
__device__ __forceinline__ int kaOf(int j, int halfk) {
  return j + 16 * halfk;
}
union V16U { struct { uint4 lo, hi; } q; v16bf v; };

__device__ __forceinline__ float blockReduceSum256(float v, float* red) {
  const int t = threadIdx.x;
  red[t] = v; __syncthreads();
  for (int o = 128; o > 0; o >>= 1) {
    if (t < o) red[t] += red[t + o];
    __syncthreads();
  }
  float r = red[0]; __syncthreads();
  return r;
}

// ---------------------------------------------------------------------------
// Weight pre-packs (once per launch).
// ---------------------------------------------------------------------------
__global__ void pack_weights(const float* __restrict__ w,
                             unsigned short* __restrict__ o,
                             int Cout, int Ktot, int KtotPad) {
  const int i = blockIdx.x * 256 + threadIdx.x;
  if (i >= Cout * KtotPad) return;
  const int n = i / KtotPad, k = i % KtotPad;
  o[i] = (k < Ktot) ? f2bf(w[(size_t)n * Ktot + k]) : (unsigned short)0;
}

__global__ void pack_w1d(const float* __restrict__ w,
                         unsigned short* __restrict__ o,
                         int Cout, int Cin, int Kw) {
  const int i = blockIdx.x * 256 + threadIdx.x;
  if (i >= Kw * Cout * Cin) return;
  const int kpos = i / (Cout * Cin);
  const int rem  = i % (Cout * Cin);
  const int n = rem / Cin, c = rem % Cin;
  o[i] = f2bf(w[((size_t)n * Cin + c) * Kw + kpos]);
}

// ---------------------------------------------------------------------------
// Kernel 1: fused stereo->mono 1-D conv stack, activations in LDS.
// Input window (incl. 7-sample halo) staged via CDNA5 async-to-LDS DMA.
// ---------------------------------------------------------------------------
#define S2M_CHUNK 128

__global__ __launch_bounds__(256) void s2m_fused(
    const float* __restrict__ wav,
    const unsigned short* __restrict__ w1p,   // [64][32]    im2col padded
    const float* __restrict__ b1, const float* __restrict__ g1, const float* __restrict__ be1,
    const unsigned short* __restrict__ w2p,   // [5][64][64] kw-major
    const float* __restrict__ b2, const float* __restrict__ g2, const float* __restrict__ be2,
    const unsigned short* __restrict__ w3p,   // [3][64][64] kw-major
    const float* __restrict__ b3, const float* __restrict__ g3, const float* __restrict__ be3,
    const float* __restrict__ w4, const float* __restrict__ b4,
    unsigned short* __restrict__ mono)
{
  __shared__ alignas(16) float         wavS[2 * 160];   // staged input + halo
  __shared__ alignas(32) unsigned short a1[144 * 64];
  __shared__ alignas(32) unsigned short a2[144 * 64];
  __shared__ alignas(32) unsigned short a3[128 * 64];

  const int b    = blockIdx.y;
  const int s    = blockIdx.x * S2M_CHUNK;
  const int wave = threadIdx.x >> 5;
  const int lane = threadIdx.x & 31;
  const int mrow = lane & 15;
  const int hk   = lane >> 4;
  const int ncol = lane & 15;
  const int sm7  = s - 7;                    // first staged sample index

  // ---- async DMA of the input window into LDS (ASYNCcnt path).
  // Edge addresses are clamped; the im2col gather re-applies zero padding,
  // so clamped edge values are never consumed.
  for (int i = threadIdx.x; i < 2 * 160; i += 256) {
    const int c = i / 160, p = i % 160;
    int xi = sm7 + p;
    xi = xi < 0 ? 0 : (xi > 63999 ? 63999 : xi);
    const float* src = wav + ((size_t)b * 2 + c) * 64000 + xi;
    const unsigned dst = (unsigned)(uintptr_t)&wavS[i];  // LDS byte offset
    asm volatile("global_load_async_to_lds_b32 %0, %1, off"
                 :: "v"(dst), "v"(src) : "memory");
  }
  asm volatile("s_wait_asynccnt 0" ::: "memory");
  __syncthreads();

  // ---- layer 1: Cin=2, Kw=9 (Ktot=18 im2col, padded 32)
  for (int job = wave; job < 9 * 4; job += 8) {
    const int mt = job >> 2, nt = job & 3;
    const int n = nt * 16 + ncol;
    const int r = s - 3 + mt * 16 + mrow;
    v8f acc = {};
    v16bf va;
#pragma unroll
    for (int j = 0; j < 16; ++j) {
      const int ka = kaOf(j, hk);
      float av = 0.f;
      if (ka < 18) {
        const int c = ka / 9, kw = ka % 9;
        const int xi = r - 4 + kw;
        if (xi >= 0 && xi < 64000) av = wavS[c * 160 + (xi - sm7)];
      }
      va[j] = (__bf16)av;
    }
    const v16bf vb = *(const v16bf*)(w1p + n * 32 + 16 * hk);
    acc = WMMA_BF16(va, vb, acc);
    const float gg = g1[n], bb = be1[n], bi = b1[n];
#pragma unroll
    for (int r8 = 0; r8 < 8; ++r8) {
      const int slot = mt * 16 + r8 + 8 * hk;
      float v = (acc[r8] + bi) * gg + bb;
      v = v > 0.f ? v : 0.f;
      if (slot < 144) a1[slot * 64 + n] = f2bf(v);
    }
  }
  __syncthreads();

  // ---- layer 2: Kw=5, channels-as-K
  for (int job = wave; job < 9 * 4; job += 8) {
    const int mt = job >> 2, nt = job & 3;
    const int n = nt * 16 + ncol;
    const int slotbase = mt * 16 + mrow;
    v8f acc = {};
#pragma unroll
    for (int kw = 0; kw < 5; ++kw) {
      int slot = slotbase + kw;
      if (slot > 143) slot = 143;
      const unsigned short* ar = a1 + slot * 64;
      const unsigned short* br = w2p + (kw * 64 + n) * 64;
#pragma unroll
      for (int kc = 0; kc < 2; ++kc) {
        V16U a;
        a.q.lo = *(const uint4*)(ar + kc * 32 + 16 * hk);
        a.q.hi = *(const uint4*)(ar + kc * 32 + 16 * hk + 8);
        const v16bf vb = *(const v16bf*)(br + kc * 32 + 16 * hk);
        acc = WMMA_BF16(a.v, vb, acc);
      }
    }
    const float gg = g2[n], bb = be2[n], bi = b2[n];
#pragma unroll
    for (int r8 = 0; r8 < 8; ++r8) {
      const int slot = mt * 16 + r8 + 8 * hk;
      float v = (acc[r8] + bi) * gg + bb;
      v = v > 0.f ? v : 0.f;
      if (slot < 144) a2[slot * 64 + n] = f2bf(v);
    }
  }
  __syncthreads();

  // ---- layer 3: Kw=3, channels-as-K
  for (int job = wave; job < 8 * 4; job += 8) {
    const int mt = job >> 2, nt = job & 3;
    const int n = nt * 16 + ncol;
    const int slotbase = mt * 16 + mrow;
    v8f acc = {};
#pragma unroll
    for (int kw = 0; kw < 3; ++kw) {
      const unsigned short* ar = a2 + (slotbase + kw) * 64;
      const unsigned short* br = w3p + (kw * 64 + n) * 64;
#pragma unroll
      for (int kc = 0; kc < 2; ++kc) {
        V16U a;
        a.q.lo = *(const uint4*)(ar + kc * 32 + 16 * hk);
        a.q.hi = *(const uint4*)(ar + kc * 32 + 16 * hk + 8);
        const v16bf vb = *(const v16bf*)(br + kc * 32 + 16 * hk);
        acc = WMMA_BF16(a.v, vb, acc);
      }
    }
    const float gg = g3[n], bb = be3[n], bi = b3[n];
#pragma unroll
    for (int r8 = 0; r8 < 8; ++r8) {
      const int slot = mt * 16 + r8 + 8 * hk;
      float v = (acc[r8] + bi) * gg + bb;
      v = v > 0.f ? v : 0.f;
      a3[slot * 64 + n] = f2bf(v);
    }
  }
  __syncthreads();

  // ---- layer 4: Cout=1, k=1, tanh
  if (threadIdx.x < S2M_CHUNK) {
    const int p = threadIdx.x;
    float sum = b4[0];
#pragma unroll 8
    for (int c = 0; c < 64; ++c) sum += bf2f(a3[p * 64 + c]) * w4[c];
    mono[(size_t)b * 64000 + s + p] = f2bf(tanhf(sum));
  }
}

// ---------------------------------------------------------------------------
// Kernel 2: cochleagram GEMM (M=6368, N=128, K=400 padded 416).
// Each wave: one M-tile x two N-tiles (shared A fragment).
// ---------------------------------------------------------------------------
__global__ __launch_bounds__(128) void coch_wmma(
    const unsigned short* __restrict__ mono,
    const unsigned short* __restrict__ fbp,
    unsigned short* __restrict__ coch)
{
  const int wave = threadIdx.x >> 5;
  const int mt   = blockIdx.x * 4 + wave;
  if (mt >= 398) return;
  const int lane = threadIdx.x & 31;
  const int mrow = lane & 15, hk = lane >> 4;
  const int n0   = blockIdx.y * 32 + (lane & 15);

  const int m0 = mt * 16 + mrow;
  const int bb0 = m0 / 398, tt0 = m0 % 398;
  const unsigned short* arow  = mono + (size_t)bb0 * 64000 + tt0 * 160;
  const unsigned short* brow0 = fbp + (size_t)n0 * 416;
  const unsigned short* brow1 = fbp + (size_t)(n0 + 16) * 416;

  v8f acc0 = {}, acc1 = {};
#pragma unroll
  for (int kc = 0; kc < 12; ++kc) {
    V16U a;
    a.q.lo = *(const uint4*)(arow + kc * 32 + 16 * hk);
    a.q.hi = *(const uint4*)(arow + kc * 32 + 16 * hk + 8);
    const v16bf vb0 = *(const v16bf*)(brow0 + kc * 32 + 16 * hk);
    const v16bf vb1 = *(const v16bf*)(brow1 + kc * 32 + 16 * hk);
    acc0 = WMMA_BF16(a.v, vb0, acc0);
    acc1 = WMMA_BF16(a.v, vb1, acc1);
  }
  {
    v16bf va;
#pragma unroll
    for (int j = 0; j < 16; ++j) {
      const int ka = 384 + kaOf(j, hk);
      va[j] = bfbits(ka < 400 ? arow[ka] : (unsigned short)0);
    }
    const v16bf vb0 = *(const v16bf*)(brow0 + 384 + 16 * hk);
    const v16bf vb1 = *(const v16bf*)(brow1 + 384 + 16 * hk);
    acc0 = WMMA_BF16(va, vb0, acc0);
    acc1 = WMMA_BF16(va, vb1, acc1);
  }
#pragma unroll
  for (int r8 = 0; r8 < 8; ++r8) {
    const int m = mt * 16 + r8 + 8 * hk;
    const int bb = m / 398, t = m % 398;
    const float v0 = powf(fabsf(acc0[r8]) + 1e-6f, 0.3f);
    const float v1 = powf(fabsf(acc1[r8]) + 1e-6f, 0.3f);
    coch[((size_t)bb * 128 + n0) * 398 + t]      = f2bf(v0);
    coch[((size_t)bb * 128 + n0 + 16) * 398 + t] = f2bf(v1);
  }
}

// ---------------------------------------------------------------------------
// Kernel 3a: first conv layer (Cin small, 7x7 s2): im2col, LDS-staged A tile
// shared by 4 waves.  Output NHWC bf16.
// ---------------------------------------------------------------------------
template <bool INF32>
__global__ __launch_bounds__(128) void conv7_first_wmma(
    const void* __restrict__ xin, const unsigned short* __restrict__ wbf,
    unsigned short* __restrict__ yout,
    int B, int Cin, int H, int W, int Cout, int Ho, int Wo, int pad)
{
  __shared__ alignas(32) unsigned short As[16 * 32];

  const int t    = threadIdx.x;
  const int wave = t >> 5;
  const int lane = t & 31;
  const int mt   = blockIdx.x;
  const int nt   = blockIdx.y * 4 + wave;

  const int Ktot   = Cin * 49;
  const int nchunk = (Ktot + 31) >> 5;
  const int KtotP  = nchunk * 32;
  const int HoWo   = Ho * Wo;

  const int sl  = t >> 2;
  const int j0  = (t & 3) * 4;
  const int shk = sl >> 4;
  const int m   = mt * 16 + (sl & 15);
  const int bb  = m / HoWo;
  const int r2  = m % HoWo;
  const int oh  = r2 / Wo, ow = r2 % Wo;
  const int ih0 = oh * 2 - pad, iw0 = ow * 2 - pad;
  const unsigned short* xb = (const unsigned short*)xin;
  const float*          xf = (const float*)xin;
  const size_t inBase = (size_t)bb * Cin;

  const int n = nt * 16 + (lane & 15);
  const unsigned short* wrow = wbf + (size_t)n * KtotP;

  v8f acc = {};
  for (int kc = 0; kc < nchunk; ++kc) {
    unsigned short e[4];
#pragma unroll
    for (int q = 0; q < 4; ++q) {
      const int j = j0 + q;
      int ka = kc * 32 + kaOf(j, shk);
      if (ka >= Ktot) ka = Ktot - 1;
      const int c = ka / 49, rem = ka % 49;
      const int kh = rem / 7, kw = rem % 7;
      const int ih = ih0 + kh, iw = iw0 + kw;
      unsigned short v = 0;
      if (ih >= 0 && ih < H && iw >= 0 && iw < W) {
        const size_t off = ((inBase + c) * H + ih) * W + iw;
        v = INF32 ? f2bf(xf[off]) : xb[off];
      }
      e[q] = v;
    }
    uint2 packed;
    packed.x = (unsigned)e[0] | ((unsigned)e[1] << 16);
    packed.y = (unsigned)e[2] | ((unsigned)e[3] << 16);
    *(uint2*)(As + sl * 16 + j0) = packed;
    __syncthreads();

    const v16bf va = *(const v16bf*)(As + lane * 16);
    const v16bf vb = *(const v16bf*)(wrow + kc * 32 + 16 * (lane >> 4));
    acc = WMMA_BF16(va, vb, acc);
    __syncthreads();
  }

  const int hkl = lane >> 4;
#pragma unroll
  for (int r8 = 0; r8 < 8; ++r8) {
    const int mm = mt * 16 + r8 + 8 * hkl;
    float v = acc[r8];
    v = v > 0.f ? v : 0.f;
    yout[(size_t)mm * Cout + n] = f2bf(v);     // NHWC
  }
}

// ---------------------------------------------------------------------------
// Kernel 3b: 3x3 s2 conv, NHWC bf16, (kh,kw)-decomposed implicit GEMM.
// Each wave: one M-tile x TWO N-tiles sharing one A fragment.
// ---------------------------------------------------------------------------
template <bool OUTF32>
__global__ __launch_bounds__(128) void conv3x3_nhwc_wmma(
    const unsigned short* __restrict__ x, const unsigned short* __restrict__ wr,
    void* __restrict__ yout,
    int B, int Cin, int H, int W, int Cout, int Ho, int Wo)
{
  const int wave = threadIdx.x >> 5;
  const int lane = threadIdx.x & 31;
  const int mt   = blockIdx.x;
  const int mrow = lane & 15, hk = lane >> 4;
  const int n0   = (blockIdx.y * 4 + wave) * 32 + (lane & 15);

  const int HoWo = Ho * Wo;
  const int m  = mt * 16 + mrow;
  const int bb = m / HoWo, r2 = m % HoWo;
  const int oh = r2 / Wo, ow = r2 % Wo;
  const int ih0 = oh * 2 - 1, iw0 = ow * 2 - 1;
  const int nkc = Cin >> 5;

  const uint4 z4 = make_uint4(0u, 0u, 0u, 0u);
  v8f acc0 = {}, acc1 = {};
#pragma unroll
  for (int kh = 0; kh < 3; ++kh) {
    const int ih = ih0 + kh;
    const bool vh = (unsigned)ih < (unsigned)H;
#pragma unroll
    for (int kw = 0; kw < 3; ++kw) {
      const int iw = iw0 + kw;
      const bool ok = vh && ((unsigned)iw < (unsigned)W);
      const unsigned short* ar =
          x + ((size_t)(bb * H + (ok ? ih : 0)) * W + (ok ? iw : 0)) * Cin;
      const unsigned short* br = wr + ((size_t)(kh * 3 + kw) * Cout + n0) * Cin;
      __builtin_prefetch(br + (size_t)16 * Cin, 0, 1);  // global_prefetch_b8
      for (int kc = 0; kc < nkc; ++kc) {
        V16U a;
        if (ok) {
          a.q.lo = *(const uint4*)(ar + kc * 32 + 16 * hk);
          a.q.hi = *(const uint4*)(ar + kc * 32 + 16 * hk + 8);
        } else {
          a.q.lo = z4; a.q.hi = z4;
        }
        const v16bf vb0 = *(const v16bf*)(br + kc * 32 + 16 * hk);
        const v16bf vb1 = *(const v16bf*)(br + (size_t)16 * Cin + kc * 32 + 16 * hk);
        acc0 = WMMA_BF16(a.v, vb0, acc0);
        acc1 = WMMA_BF16(a.v, vb1, acc1);
      }
    }
  }
#pragma unroll
  for (int r8 = 0; r8 < 8; ++r8) {
    const int mm = mt * 16 + r8 + 8 * hk;
    float v0 = acc0[r8], v1 = acc1[r8];
    v0 = v0 > 0.f ? v0 : 0.f;
    v1 = v1 > 0.f ? v1 : 0.f;
    const size_t off = (size_t)mm * Cout + n0;  // NHWC
    if (OUTF32) {
      ((float*)yout)[off]      = v0;
      ((float*)yout)[off + 16] = v1;
    } else {
      ((unsigned short*)yout)[off]      = f2bf(v0);
      ((unsigned short*)yout)[off + 16] = f2bf(v1);
    }
  }
}

// ---------------------------------------------------------------------------
// Tail kernels (NHWC layouts; channel loops unit-stride)
// ---------------------------------------------------------------------------
__global__ void l2norm_nhwc(float* __restrict__ x, int C) {
  const size_t pos = blockIdx.x;
  float* row = x + pos * C;
  __shared__ float red[64];
  float s = 0.f;
  for (int c = threadIdx.x; c < C; c += 64) { const float v = row[c]; s += v * v; }
  red[threadIdx.x] = s; __syncthreads();
  for (int o = 32; o > 0; o >>= 1) {
    if (threadIdx.x < o) red[threadIdx.x] += red[threadIdx.x + o];
    __syncthreads();
  }
  float nrm = sqrtf(red[0]);
  nrm = nrm < 1e-12f ? 1e-12f : nrm;
  const float inv = 1.f / nrm;
  for (int c = threadIdx.x; c < C; c += 64) row[c] *= inv;
}

__global__ void maxpool_nhwc(const float* __restrict__ x, float* __restrict__ y,
                             int C, int HW) {
  const int nc = blockIdx.x;
  const int nb = nc / C, c = nc % C;
  __shared__ float red[64];
  float m = -3.4e38f;
  for (int i = threadIdx.x; i < HW; i += 64) {
    const float v = x[((size_t)nb * HW + i) * C + c];
    m = v > m ? v : m;
  }
  red[threadIdx.x] = m; __syncthreads();
  for (int o = 32; o > 0; o >>= 1) {
    if (threadIdx.x < o) {
      const float b = red[threadIdx.x + o];
      if (b > red[threadIdx.x]) red[threadIdx.x] = b;
    }
    __syncthreads();
  }
  if (threadIdx.x == 0) y[nc] = red[0];
}

__global__ __launch_bounds__(256) void l2norm_vec(float* __restrict__ x, int C) {
  __shared__ float red[256];
  const int n = blockIdx.x;
  float s = 0.f;
  for (int c = threadIdx.x; c < C; c += 256) { const float v = x[(size_t)n * C + c]; s += v * v; }
  s = blockReduceSum256(s, red);
  float nrm = sqrtf(s);
  nrm = nrm < 1e-12f ? 1e-12f : nrm;
  const float inv = 1.f / nrm;
  for (int c = threadIdx.x; c < C; c += 256) x[(size_t)n * C + c] *= inv;
}

__global__ __launch_bounds__(256) void attn_kernel(
    const float* __restrict__ imgN,            // NHWC
    const float* __restrict__ audvec,
    float* __restrict__ simgrid, float* __restrict__ sim12,
    float* __restrict__ outA, float* __restrict__ outPos, float* __restrict__ outNeg)
{
  const int n = blockIdx.x, k = blockIdx.y;
  __shared__ float av[512];
  __shared__ float red[256];
  const int t = threadIdx.x;
  for (int c = t; c < 512; c += 256) av[c] = audvec[k * 512 + c];
  __syncthreads();

  float a0 = 0.f;
  if (t < 196) {
    const float* ip = imgN + ((size_t)n * 196 + t) * 512;
#pragma unroll 4
    for (int c = 0; c < 512; ++c) a0 += ip[c] * av[c];
  }
  const float it = 1.f / 0.03f;
  const float valid = (t < 196) ? 1.f : 0.f;
  const float posall = valid / (1.f + expf(-(a0 - 0.65f) * it));

  const float S1 = blockReduceSum256(posall * a0, red);
  const float S2 = blockReduceSum256(posall, red);
  if (t == 0) simgrid[n * 16 + k] = S1 / S2;

  if (n == k) {
    const float pos  = posall;
    const float pos2 = valid / (1.f + expf(-(a0 - 0.40f) * it));
    const float neg  = valid - pos2;
    if (t < 196) {
      outA[n * 196 + t]   = a0;
      outPos[n * 196 + t] = pos;
      outNeg[n * 196 + t] = neg;
    }
    const float P1 = blockReduceSum256(pos * a0, red);
    const float P2 = blockReduceSum256(pos, red);
    const float N1 = blockReduceSum256(neg * a0, red);
    const float N2 = blockReduceSum256(neg, red);
    if (t == 0) { sim12[n * 2] = P1 / P2; sim12[n * 2 + 1] = N1 / N2; }
  }
}

__global__ void logits_kernel(const float* __restrict__ simgrid,
                              const float* __restrict__ sim12,
                              float* __restrict__ outL)
{
  const int n = blockIdx.x;
  const int j = threadIdx.x;
  if (j >= 18) return;
  float v;
  if (j == 0)       v = sim12[n * 2];
  else if (j < 17)  { const int k = j - 1; v = simgrid[n * 16 + k] * (k == n ? -99.f : 1.f); }
  else              v = sim12[n * 2 + 1];
  outL[n * 18 + j] = v * (1.f / 0.07f);
}

// ---------------------------------------------------------------------------
// Host launcher
// ---------------------------------------------------------------------------
static inline int grid256(int total) { return (total + 255) / 256; }

extern "C" void kernel_launch(void* const* d_in, const int* in_sizes, int n_in,
                              void* d_out, int out_size, void* d_ws, size_t ws_size,
                              hipStream_t stream) {
  (void)in_sizes; (void)n_in; (void)out_size; (void)ws_size;

  const float* image  = (const float*)d_in[0];
  const float* wav    = (const float*)d_in[1];
  const float* s_w1   = (const float*)d_in[2];
  const float* s_b1   = (const float*)d_in[3];
  const float* s_g1   = (const float*)d_in[4];
  const float* s_be1  = (const float*)d_in[5];
  const float* s_w2   = (const float*)d_in[6];
  const float* s_b2   = (const float*)d_in[7];
  const float* s_g2   = (const float*)d_in[8];
  const float* s_be2  = (const float*)d_in[9];
  const float* s_w3   = (const float*)d_in[10];
  const float* s_b3   = (const float*)d_in[11];
  const float* s_g3   = (const float*)d_in[12];
  const float* s_be3  = (const float*)d_in[13];
  const float* s_w4   = (const float*)d_in[14];
  const float* s_b4   = (const float*)d_in[15];
  const float* vis0   = (const float*)d_in[16];
  const float* vis1   = (const float*)d_in[17];
  const float* vis2   = (const float*)d_in[18];
  const float* vis3   = (const float*)d_in[19];
  const float* aud0   = (const float*)d_in[20];
  const float* aud1   = (const float*)d_in[21];
  const float* aud2   = (const float*)d_in[22];
  const float* aud3   = (const float*)d_in[23];
  const float* fb     = (const float*)d_in[24];

  char* ws = (char*)d_ws;
  const size_t MB = (size_t)1 << 20, KB = 1024;
  unsigned short* mono = (unsigned short*)(ws + 0);
  unsigned short* coch = (unsigned short*)(ws + 2 * MB);
  unsigned short* p0   = (unsigned short*)(ws + 4 * MB);     // NHWC ping
  unsigned short* p1   = (unsigned short*)(ws + 32 * MB);    // NHWC pong
  float* a4f     = (float*)(ws + 48 * MB);                   // NHWC f32
  float* v4f     = (float*)(ws + 56 * MB);                   // NHWC f32
  float* avec    = (float*)(ws + 63 * MB);
  float* simgrid = (float*)(ws + 63 * MB + 64 * KB);
  float* sim12   = (float*)(ws + 63 * MB + 68 * KB);
  unsigned short* wv0p = (unsigned short*)(ws + 72 * MB);
  unsigned short* wv1p = (unsigned short*)(ws + 72 * MB + 64 * KB);
  unsigned short* wv2p = (unsigned short*)(ws + 72 * MB + 256 * KB);
  unsigned short* wv3p = (unsigned short*)(ws + 73 * MB);
  unsigned short* wa0p = (unsigned short*)(ws + 76 * MB);
  unsigned short* wa1p = (unsigned short*)(ws + 76 * MB + 64 * KB);
  unsigned short* wa2p = (unsigned short*)(ws + 76 * MB + 256 * KB);
  unsigned short* wa3p = (unsigned short*)(ws + 77 * MB);
  unsigned short* ws1p = (unsigned short*)(ws + 80 * MB);
  unsigned short* ws2p = (unsigned short*)(ws + 80 * MB + 64 * KB);
  unsigned short* ws3p = (unsigned short*)(ws + 80 * MB + 128 * KB);
  unsigned short* fbp  = (unsigned short*)(ws + 80 * MB + 192 * KB);

  float* out    = (float*)d_out;
  float* outA   = out;
  float* outL   = out + 3136;
  float* outPos = out + 3424;
  float* outNeg = out + 6560;

  // ---- weight pre-packs
  pack_weights<<<grid256(64 * 32),   256, 0, stream>>>(s_w1, ws1p, 64, 18, 32);
  pack_w1d   <<<grid256(5 * 64 * 64),  256, 0, stream>>>(s_w2, ws2p, 64, 64, 5);
  pack_w1d   <<<grid256(3 * 64 * 64),  256, 0, stream>>>(s_w3, ws3p, 64, 64, 3);
  pack_weights<<<grid256(128 * 416), 256, 0, stream>>>(fb, fbp, 128, 400, 416);
  pack_weights<<<grid256(64 * 160),  256, 0, stream>>>(vis0, wv0p, 64, 147, 160);
  pack_w1d   <<<grid256(9 * 128 * 64),  256, 0, stream>>>(vis1, wv1p, 128, 64, 9);
  pack_w1d   <<<grid256(9 * 256 * 128), 256, 0, stream>>>(vis2, wv2p, 256, 128, 9);
  pack_w1d   <<<grid256(9 * 512 * 256), 256, 0, stream>>>(vis3, wv3p, 512, 256, 9);
  pack_weights<<<grid256(64 * 64),   256, 0, stream>>>(aud0, wa0p, 64, 49, 64);
  pack_w1d   <<<grid256(9 * 128 * 64),  256, 0, stream>>>(aud1, wa1p, 128, 64, 9);
  pack_w1d   <<<grid256(9 * 256 * 128), 256, 0, stream>>>(aud2, wa2p, 256, 128, 9);
  pack_w1d   <<<grid256(9 * 512 * 256), 256, 0, stream>>>(aud3, wa3p, 512, 256, 9);

  // ---- audio: fused s2m -> mono (bf16)
  s2m_fused<<<dim3(64000 / S2M_CHUNK, 16), 256, 0, stream>>>(
      wav, ws1p, s_b1, s_g1, s_be1, ws2p, s_b2, s_g2, s_be2,
      ws3p, s_b3, s_g3, s_be3, s_w4, s_b4, mono);

  // ---- cochleagram
  coch_wmma<<<dim3(100, 4), 128, 0, stream>>>(mono, fbp, coch);

  // ---- audio backbone (NHWC after first layer)
  conv7_first_wmma<false><<<dim3(12736, 1), 128, 0, stream>>>(coch, wa0p, p0, 16, 1, 128, 398, 64, 64, 199, 3);
  conv3x3_nhwc_wmma<false><<<dim3(3200, 1), 128, 0, stream>>>(p0, wa1p, p1, 16,  64,  64, 199, 128, 32, 100);
  conv3x3_nhwc_wmma<false><<<dim3(800,  2), 128, 0, stream>>>(p1, wa2p, p0, 16, 128,  32, 100, 256, 16,  50);
  conv3x3_nhwc_wmma<true ><<<dim3(200,  4), 128, 0, stream>>>(p0, wa3p, a4f,16, 256,  16,  50, 512,  8,  25);

  maxpool_nhwc<<<16 * 512, 64, 0, stream>>>(a4f, avec, 512, 200);
  l2norm_vec<<<16, 256, 0, stream>>>(avec, 512);

  // ---- visual backbone (NHWC after first layer)
  conv7_first_wmma<true><<<dim3(12544, 1), 128, 0, stream>>>(image, wv0p, p0, 16, 3, 224, 224, 64, 112, 112, 3);
  conv3x3_nhwc_wmma<false><<<dim3(3136, 1), 128, 0, stream>>>(p0, wv1p, p1, 16,  64, 112, 112, 128, 56, 56);
  conv3x3_nhwc_wmma<false><<<dim3(784,  2), 128, 0, stream>>>(p1, wv2p, p0, 16, 128,  56,  56, 256, 28, 28);
  conv3x3_nhwc_wmma<true ><<<dim3(196,  4), 128, 0, stream>>>(p0, wv3p, v4f,16, 256,  28,  28, 512, 14, 14);

  l2norm_nhwc<<<16 * 196, 64, 0, stream>>>(v4f, 512);

  // ---- attention + logits
  attn_kernel<<<dim3(16, 16), 256, 0, stream>>>(v4f, avec, simgrid, sim12,
                                                outA, outPos, outNeg);
  logits_kernel<<<16, 32, 0, stream>>>(simgrid, sim12, outL);
}
